// SpacetimeRegularizedModel_48438641164406
// MI455X (gfx1250) — compile-verified, matrix operand order checked by reading, not run on hardware
//
#include <hip/hip_runtime.h>

// CDNA5 / gfx1250, wave32. WMMA vector operand types.
typedef __attribute__((ext_vector_type(16))) _Float16 v16h;
typedef __attribute__((ext_vector_type(8)))  float    v8f;
typedef __attribute__((ext_vector_type(4)))  float    v4f;

#define SEQ        1024      // S
#define NLAYERS    12        // L
#define LAMBDA_REG 0.01f
#define CHUNK      512       // floats processed per wave per iteration (one WMMA A tile)

// ---------------------------------------------------------------------------
// Zero the single-float accumulator in workspace (d_ws is poisoned by harness).
// ---------------------------------------------------------------------------
__global__ void st_init(float* __restrict__ acc) {
    acc[0] = 0.0f;
}

// ---------------------------------------------------------------------------
// Streaming masked-square reduction.
//
// Each wave grid-strides over 512-float chunks. A chunk is row-aligned inside
// an S=1024-long row, so the attention row index i = (base/1024) % 1024 and the
// lane's column base j0 = base % 1024 + lane*16 are shift/and.
//
// Per chunk: lane loads 16 consecutive f32 (4 x b128), computes
//   v = a*a * ((-dt*dt) > 0 ? 1 : 0)        (mask is the faithful spacelike test)
// packs the 16 values into a v16h A-tile, and accumulates
//   C += A x Ones(32x16)  via v_wmma_f32_16x16x32_f16.
// Sum over all entries of D equals 16 * sum(A) independent of packing order,
// so any lane->(M,K) assignment is valid for a global sum.
// ---------------------------------------------------------------------------
__global__ __launch_bounds__(256) void st_reduce(const float* __restrict__ attn,
                                                 float* __restrict__ acc,
                                                 long long nChunks) {
    const int  lane        = threadIdx.x & 31;
    const int  waveInBlk   = threadIdx.x >> 5;
    const long long wavesPerBlk = blockDim.x >> 5;
    const long long wave        = (long long)blockIdx.x * wavesPerBlk + waveInBlk;
    const long long totalWaves  = (long long)gridDim.x * wavesPerBlk;

    v8f c = {};            // f32 16x16 accumulator tile (8 VGPRs / lane, wave32)
    v16h ones;
#pragma unroll
    for (int k = 0; k < 16; ++k) ones[k] = (_Float16)1.0f;

    for (long long ch = wave; ch < nChunks; ch += totalWaves) {
        const long long base = ch * (long long)CHUNK;       // row-aligned chunk
        const int i  = (int)((base >> 10) & (SEQ - 1));     // row index within S
        const int j0 = (int)(base & (SEQ - 1)) + lane * 16; // lane's first column

        const float* p = attn + base + (long long)lane * 16;
        // Prefetch the next grid-stride chunk (speculative, drops on OOB).
        __builtin_prefetch(p + totalWaves * (long long)CHUNK, 0, 0);

        const v4f x0 = *(const v4f*)(p + 0);
        const v4f x1 = *(const v4f*)(p + 4);
        const v4f x2 = *(const v4f*)(p + 8);
        const v4f x3 = *(const v4f*)(p + 12);

        v16h a;
#pragma unroll
        for (int e = 0; e < 16; ++e) {
            const float x = (e < 4) ? x0[e] : (e < 8) ? x1[e - 4]
                          : (e < 12) ? x2[e - 8] : x3[e - 12];
            int dt = (j0 + e) - i;
            dt = dt < 0 ? -dt : dt;
            const int   s2 = -(dt * dt);                 // dx = 0
            const float m  = (s2 > 0) ? 1.0f : 0.0f;     // spacelike mask
            a[e] = (_Float16)(x * x * m);                // exact 0.0 when masked
        }

        // D = A x Ones + C  -> accumulate running sums on the matrix pipe.
        c = __builtin_amdgcn_wmma_f32_16x16x32_f16(
                /*neg_a=*/false, a, /*neg_b=*/false, ones,
                /*c_mod=*/(short)0, c, /*reuse_a=*/false, /*reuse_b=*/false);
    }

    // Per-lane sum of the C tile, then wave32 shuffle reduction.
    float s = c[0] + c[1] + c[2] + c[3] + c[4] + c[5] + c[6] + c[7];
#pragma unroll
    for (int off = 16; off > 0; off >>= 1)
        s += __shfl_xor(s, off, 32);

    if (lane == 0)
        atomicAdd(acc, s * (1.0f / 16.0f));   // undo the x16 from B = ones
}

// ---------------------------------------------------------------------------
// total_loss = base_loss + LAMBDA * (penalty / L)
// ---------------------------------------------------------------------------
__global__ void st_final(const float* __restrict__ base_loss,
                         const float* __restrict__ acc,
                         float* __restrict__ out) {
    out[0] = base_loss[0] + LAMBDA_REG * (acc[0] / (float)NLAYERS);
}

extern "C" void kernel_launch(void* const* d_in, const int* in_sizes, int n_in,
                              void* d_out, int out_size, void* d_ws, size_t ws_size,
                              hipStream_t stream) {
    const float* attn      = (const float*)d_in[0];  // [L,B,H,S,S] f32
    const float* base_loss = (const float*)d_in[1];  // scalar f32
    float*       out       = (float*)d_out;          // scalar f32
    float*       acc       = (float*)d_ws;           // 1-float accumulator

    const long long N       = (long long)in_sizes[0];   // 301,989,888 (mult. of 512)
    const long long nChunks = N / CHUNK;

    st_init<<<1, 1, 0, stream>>>(acc);

    // 1728 blocks x 8 wave32 = 13,824 waves; ~43 chunks each -> HBM-saturating
    // grid-stride with enough latency hiding per WGP.
    st_reduce<<<1728, 256, 0, stream>>>(attn, acc, nChunks);

    st_final<<<1, 1, 0, stream>>>(base_loss, acc, out);
}
